// SparseMatrixEquivariantNetwork_59210419142985
// MI455X (gfx1250) — matrix operand order repaired and checked by reading, not compiled, not run on hardware
//
#include <hip/hip_runtime.h>

// ---------------------------------------------------------------------------
// CDNA5 (gfx1250, wave32) implementation of the sparse equivariant network.
// Matmuls run on v_wmma_f32_16x16x32_bf16. RelationNorm is folded into the
// next layer's weights (the layer is linear), removing 3 normalize passes.
// ---------------------------------------------------------------------------

typedef __attribute__((ext_vector_type(16))) __bf16 v16bf;
typedef __attribute__((ext_vector_type(8)))  float  v8f;

#define WMMA_BF16(A, B, C) \
    __builtin_amdgcn_wmma_f32_16x16x32_bf16(false, (A), false, (B), (short)0, (C), false, false)

// Load one A-operand fragment (16x32 bf16, M=row per lane pair) from an f32 row.
// Layout (ISA 7.12.2, 16-bit A 16x32): lanes 0-15 hold K in {0..7, 16..23},
// lanes 16-31 hold K in {8..15, 24..31}; elem e<8 -> k=kt*32+khalf+e,
// e>=8 -> k=kt*32+16+khalf+(e-8).  khalf = (lane>=16)*8.
template<int CIN>
__device__ __forceinline__ v16bf load_a_frag(const float* __restrict__ rowp, int kt, int khalf) {
    const int k0 = kt * 32 + khalf;
    const int k1 = kt * 32 + 16 + khalf;
    float4 x0 = make_float4(0.f, 0.f, 0.f, 0.f), x1 = x0, y0 = x0, y1 = x0;
    if (k0 < CIN) {  // uniform across the wave (CIN multiple of 16, khalf in {0,8})
        const float4* p = reinterpret_cast<const float4*>(rowp + k0);
        x0 = p[0]; x1 = p[1];
    }
    if (k1 < CIN) {
        const float4* q = reinterpret_cast<const float4*>(rowp + k1);
        y0 = q[0]; y1 = q[1];
    }
    v16bf a;
    a[0] = (__bf16)x0.x; a[1] = (__bf16)x0.y; a[2]  = (__bf16)x0.z; a[3]  = (__bf16)x0.w;
    a[4] = (__bf16)x1.x; a[5] = (__bf16)x1.y; a[6]  = (__bf16)x1.z; a[7]  = (__bf16)x1.w;
    a[8] = (__bf16)y0.x; a[9] = (__bf16)y0.y; a[10] = (__bf16)y0.z; a[11] = (__bf16)y0.w;
    a[12] = (__bf16)y1.x; a[13] = (__bf16)y1.y; a[14] = (__bf16)y1.z; a[15] = (__bf16)y1.w;
    return a;
}

// ------------------------- small helper kernels ----------------------------

__global__ __launch_bounds__(256) void counts_kernel(
    const int* __restrict__ row, const int* __restrict__ col,
    float* __restrict__ rcnt, float* __restrict__ ccnt, int nnz) {
    int e = blockIdx.x * 256 + threadIdx.x;
    if (e >= nnz) return;
    atomicAdd(&rcnt[row[e]], 1.f);
    atomicAdd(&ccnt[col[e]], 1.f);
}

__global__ __launch_bounds__(256) void segsum_kernel(
    const float* __restrict__ h, const int* __restrict__ row, const int* __restrict__ col,
    float* __restrict__ rsum, float* __restrict__ csum, int nnz, int lc) {
    int total = nnz << lc;
    int t = blockIdx.x * 256 + threadIdx.x;
    if (t >= total) return;
    int e = t >> lc;
    int c = t & ((1 << lc) - 1);
    float v = h[t];
    atomicAdd(&rsum[((size_t)row[e] << lc) + c], v);
    if (csum) atomicAdd(&csum[((size_t)col[e] << lc) + c], v);
}

// In-place: rsum -> row-mean, csum -> col-mean; accumulates total sum into tmsum.
__global__ __launch_bounds__(256) void seg_finalize_kernel(
    float* __restrict__ rsum, float* __restrict__ csum,
    const float* __restrict__ rcnt, const float* __restrict__ ccnt,
    float* __restrict__ tmsum, int n, int lc) {
    __shared__ float sm[64];
    const int cin = 1 << lc;
    const int tid = threadIdx.x;
    if (tid < cin) sm[tid] = 0.f;
    __syncthreads();
    int t = blockIdx.x * 256 + tid;
    int total = n << lc;
    if (t < total) {
        int i = t >> lc, c = t & (cin - 1);
        float rv = rsum[t];
        if (tmsum) atomicAdd(&sm[c], rv);
        rsum[t] = rv / fmaxf(rcnt[i], 1.f);
        if (csum) csum[t] = csum[t] / fmaxf(ccnt[i], 1.f);
    }
    __syncthreads();
    if (tid < cin && tmsum) atomicAdd(&tmsum[tid], sm[tid]);
}

// Per-layer prep (1 block): fold previous-layer norm (s,m) into weights,
// pack bf16 B-operand tiles, and build the constant channel bias.
// B layout (matches A): lanes 0-15 hold K in {0..7, 16..23}, lanes 16-31 K in {8..15, 24..31}.
__global__ __launch_bounds__(256) void prep_kernel(
    const float* __restrict__ W,       // (7, cin, cout)
    const float* __restrict__ b,       // (cout)
    const float* __restrict__ stats,   // (2*cin) sum/sumsq of prev layer, or null
    const float* __restrict__ tmsum,   // (cin) total sum of current input
    float* __restrict__ cbias,         // (cout)
    __bf16* __restrict__ wEdge,        // [2][KT][NT][32][16]
    __bf16* __restrict__ wNode,        // [4][KT][NT][32][16]  {W2,W4,W3,W5}
    int cin, int cout, float inv_nnz) {
    __shared__ float s[64], mu[64], tmv[64];
    const int tid = threadIdx.x;
    for (int i = tid; i < cin; i += 256) {
        float mean = 0.f, sc = 1.f;
        if (stats) {
            mean = stats[i] * inv_nnz;
            float var = stats[cin + i] * inv_nnz - mean * mean;
            sc = rsqrtf(fmaxf(var, 0.f) + 1e-5f);
        }
        s[i] = sc; mu[i] = mean; tmv[i] = tmsum[i] * inv_nnz;
    }
    __syncthreads();
    for (int j = tid; j < cout; j += 256) {
        float acc = b[j];
        for (int i = 0; i < cin; ++i)
            acc += (tmv[i] - mu[i]) * s[i] * W[(6 * cin + i) * cout + j];
        for (int k = 0; k < 6; ++k)
            for (int i = 0; i < cin; ++i)
                acc -= mu[i] * s[i] * W[(k * cin + i) * cout + j];
        cbias[j] = acc;
    }
    const int KT = (cin + 31) >> 5, NT = cout >> 4;
    const int perset = KT * NT * 512;
    for (int t = tid; t < 2 * perset; t += 256) {
        int elem = t & 15, lane = (t >> 4) & 31;
        int q = t >> 9;
        int nt = q % NT; q /= NT;
        int kt = q % KT; int set = q / KT;
        int nn = lane & 15;
        int k = kt * 32 + ((lane >> 4) << 3) + (elem & 7) + ((elem >> 3) << 4);
        float v = (k < cin) ? s[k] * W[(set * cin + k) * cout + (nt << 4) + nn] : 0.f;
        wEdge[t] = (__bf16)v;
    }
    const int kmap[4] = {2, 4, 3, 5};
    for (int t = tid; t < 4 * perset; t += 256) {
        int elem = t & 15, lane = (t >> 4) & 31;
        int q = t >> 9;
        int nt = q % NT; q /= NT;
        int kt = q % KT; int set = q / KT;
        int nn = lane & 15;
        int k = kt * 32 + ((lane >> 4) << 3) + (elem & 7) + ((elem >> 3) << 4);
        float v = (k < cin) ? s[k] * W[(kmap[set] * cin + k) * cout + (nt << 4) + nn] : 0.f;
        wNode[t] = (__bf16)v;
    }
}

// ------------------------- WMMA compute kernels ----------------------------

// nodeR = rm@W2' + cm@W4', nodeC = rm@W3' + cm@W5'   (N rows)
template<int CIN, int COUT>
__global__ __launch_bounds__(256) void eq_node_kernel(
    const float* __restrict__ rm, const float* __restrict__ cm,
    const __bf16* __restrict__ wNode,
    float* __restrict__ nodeR, float* __restrict__ nodeC, int n) {
    constexpr int KT = (CIN + 31) / 32;
    constexpr int NT = COUT / 16;
    constexpr int WN = 4 * KT * NT * 512;
    __shared__ __align__(32) __bf16 sW[WN];
    {
        const int4* src = (const int4*)wNode;
        int4* dst = (int4*)sW;
        for (int i = threadIdx.x; i < WN / 8; i += 256) dst[i] = src[i];
    }
    __syncthreads();
    const int lane = threadIdx.x & 31;
    const int wave = threadIdx.x >> 5;
    const int m = lane & 15;
    const int hi = lane >> 4;
    const int khalf = hi << 3;
    const int nTiles = n >> 4;
    for (int tile = blockIdx.x * 8 + wave; tile < nTiles; tile += gridDim.x * 8) {
        const int ibase = tile << 4;
        const int im = ibase + m;
        v16bf aR[KT], aC[KT];
#pragma unroll
        for (int kt = 0; kt < KT; ++kt) {
            aR[kt] = load_a_frag<CIN>(rm + (size_t)im * CIN, kt, khalf);
            aC[kt] = load_a_frag<CIN>(cm + (size_t)im * CIN, kt, khalf);
        }
#pragma unroll
        for (int nt = 0; nt < NT; ++nt) {
            v8f cR = {0.f, 0.f, 0.f, 0.f, 0.f, 0.f, 0.f, 0.f};
            v8f cC = {0.f, 0.f, 0.f, 0.f, 0.f, 0.f, 0.f, 0.f};
#pragma unroll
            for (int kt = 0; kt < KT; ++kt) {
                v16bf b0 = *(const v16bf*)&sW[(((0 * KT + kt) * NT + nt) << 9) + (lane << 4)];
                v16bf b1 = *(const v16bf*)&sW[(((1 * KT + kt) * NT + nt) << 9) + (lane << 4)];
                v16bf b2 = *(const v16bf*)&sW[(((2 * KT + kt) * NT + nt) << 9) + (lane << 4)];
                v16bf b3 = *(const v16bf*)&sW[(((3 * KT + kt) * NT + nt) << 9) + (lane << 4)];
                cR = WMMA_BF16(aR[kt], b0, cR);
                cR = WMMA_BF16(aC[kt], b1, cR);
                cC = WMMA_BF16(aR[kt], b2, cC);
                cC = WMMA_BF16(aC[kt], b3, cC);
            }
#pragma unroll
            for (int r = 0; r < 8; ++r) {
                int i = ibase + r + (hi << 3);
                nodeR[(size_t)i * COUT + (nt << 4) + m] = cR[r];
                nodeC[(size_t)i * COUT + (nt << 4) + m] = cC[r];
            }
        }
    }
}

// Per-edge: hout = relu(v@W0' + v[perm]@W1' + nodeR[row] + nodeC[col] + cbias)
// plus per-channel sum/sumsq stats for the next layer's folded norm.
template<int CIN, int COUT>
__global__ __launch_bounds__(256) void eq_edge_kernel(
    const float* __restrict__ hin,
    const int* __restrict__ row, const int* __restrict__ col, const int* __restrict__ perm,
    const __bf16* __restrict__ wEdge,
    const float* __restrict__ nodeR, const float* __restrict__ nodeC,
    const float* __restrict__ cbias,
    float* __restrict__ hout, float* __restrict__ stats, int nnz) {
    constexpr int KT = (CIN + 31) / 32;
    constexpr int NT = COUT / 16;
    constexpr int WN = 2 * KT * NT * 512;
    __shared__ __align__(32) __bf16 sW[WN];
    {
        const int4* src = (const int4*)wEdge;
        int4* dst = (int4*)sW;
        for (int i = threadIdx.x; i < WN / 8; i += 256) dst[i] = src[i];
    }
    __syncthreads();
    const int lane = threadIdx.x & 31;
    const int wave = threadIdx.x >> 5;
    const int m = lane & 15;
    const int hi = lane >> 4;
    const int khalf = hi << 3;
    const int nTiles = nnz >> 4;
    float accS[NT], accQ[NT];
#pragma unroll
    for (int nt = 0; nt < NT; ++nt) { accS[nt] = 0.f; accQ[nt] = 0.f; }

    for (int tile = blockIdx.x * 8 + wave; tile < nTiles; tile += gridDim.x * 8) {
        const int ebase = tile << 4;
        const int em = ebase + m;
        const int pm = perm[em];
        v16bf aId[KT], aPm[KT];
#pragma unroll
        for (int kt = 0; kt < KT; ++kt) {
            aId[kt] = load_a_frag<CIN>(hin + (size_t)em * CIN, kt, khalf);
            aPm[kt] = load_a_frag<CIN>(hin + (size_t)pm * CIN, kt, khalf);
        }
#pragma unroll
        for (int nt = 0; nt < NT; ++nt) {
            v8f c = {0.f, 0.f, 0.f, 0.f, 0.f, 0.f, 0.f, 0.f};
#pragma unroll
            for (int kt = 0; kt < KT; ++kt) {
                v16bf b0 = *(const v16bf*)&sW[(((0 * KT + kt) * NT + nt) << 9) + (lane << 4)];
                v16bf b1 = *(const v16bf*)&sW[(((1 * KT + kt) * NT + nt) << 9) + (lane << 4)];
                c = WMMA_BF16(aId[kt], b0, c);
                c = WMMA_BF16(aPm[kt], b1, c);
            }
            const int ch = (nt << 4) + m;
            const float cb = cbias[ch];
            float psum = 0.f, psq = 0.f;
#pragma unroll
            for (int r = 0; r < 8; ++r) {
                const int er = ebase + r + (hi << 3);
                const int ri = row[er];
                const int ci = col[er];
                float v = c[r] + nodeR[(size_t)ri * COUT + ch]
                               + nodeC[(size_t)ci * COUT + ch] + cb;
                v = fmaxf(v, 0.f);
                hout[(size_t)er * COUT + ch] = v;
                psum += v; psq += v * v;
            }
            accS[nt] += psum; accQ[nt] += psq;
        }
    }
    // combine lane L with L+16 (same channel), one atomic per channel per wave
#pragma unroll
    for (int nt = 0; nt < NT; ++nt) {
        float s2 = accS[nt] + __shfl_xor(accS[nt], 16);
        float q2 = accQ[nt] + __shfl_xor(accQ[nt], 16);
        if (lane < 16) {
            atomicAdd(&stats[(nt << 4) + m], s2);
            atomicAdd(&stats[COUT + (nt << 4) + m], q2);
        }
    }
}

// Fold final norm into the two dense heads: Wf = diag(s3)Wp @ Wl, bf folded bias.
__global__ __launch_bounds__(256) void final_prep_kernel(
    const float* __restrict__ stats3, const float* __restrict__ Wp, const float* __restrict__ bp,
    const float* __restrict__ Wl, const float* __restrict__ bl,
    float* __restrict__ Wf, float* __restrict__ bfv, float inv_nnz) {
    __shared__ float s3[32], m3[32], t1[16];
    const int tid = threadIdx.x;
    if (tid < 32) {
        float mean = stats3[tid] * inv_nnz;
        float var = stats3[32 + tid] * inv_nnz - mean * mean;
        s3[tid] = rsqrtf(fmaxf(var, 0.f) + 1e-5f);
        m3[tid] = mean;
    }
    __syncthreads();
    if (tid < 16) {
        float acc = bp[tid];
        for (int k = 0; k < 32; ++k) acc -= m3[k] * s3[k] * Wp[k * 16 + tid];
        t1[tid] = acc;
    }
    __syncthreads();
    for (int idx = tid; idx < 512; idx += 256) {
        int k = idx >> 4, j = idx & 15;
        float acc = 0.f;
        for (int t = 0; t < 16; ++t) acc += s3[k] * Wp[k * 16 + t] * Wl[t * 16 + j];
        Wf[idx] = acc;
    }
    if (tid < 16) {
        float acc = bl[tid];
        for (int t = 0; t < 16; ++t) acc += t1[t] * Wl[t * 16 + tid];
        bfv[tid] = acc;
    }
}

// out[i] = rowmean3[i] @ Wf + bf
__global__ __launch_bounds__(256) void final_out_kernel(
    const float* __restrict__ rm3, const float* __restrict__ Wf, const float* __restrict__ bfv,
    float* __restrict__ out, int n) {
    __shared__ float sWf[512];
    __shared__ float sb[16];
    for (int i = threadIdx.x; i < 512; i += 256) sWf[i] = Wf[i];
    if (threadIdx.x < 16) sb[threadIdx.x] = bfv[threadIdx.x];
    __syncthreads();
    int i = blockIdx.x * 256 + threadIdx.x;
    if (i >= n) return;
    float acc[16];
#pragma unroll
    for (int j = 0; j < 16; ++j) acc[j] = sb[j];
    for (int k = 0; k < 32; ++k) {
        float rv = rm3[(size_t)i * 32 + k];
#pragma unroll
        for (int j = 0; j < 16; ++j) acc[j] += rv * sWf[k * 16 + j];
    }
#pragma unroll
    for (int j = 0; j < 16; ++j) out[(size_t)i * 16 + j] = acc[j];
}

// ------------------------------- host side ---------------------------------

namespace {
struct Ws {
    float *hA, *hB, *rsum, *csum, *nodeR, *nodeC, *rcnt, *ccnt, *tmsum;
    float *stats1, *stats2, *stats3, *cbias, *Wf, *bfv;
    __bf16 *wEdge, *wNode;
};
static inline int cdiv(long long a, long long b) { return (int)((a + b - 1) / b); }

template<int CIN, int COUT>
static void run_layer(const float* hin, const float* W, const float* b,
                      const float* statsPrev, float* statsOut, float* hout,
                      const Ws& ws, const int* row, const int* col, const int* perm,
                      int n, int nnz, float inv_nnz, hipStream_t stream) {
    constexpr int LC = (CIN == 16) ? 4 : ((CIN == 32) ? 5 : 6);
    hipMemsetAsync(ws.rsum, 0, (size_t)n * CIN * sizeof(float), stream);
    hipMemsetAsync(ws.csum, 0, (size_t)n * CIN * sizeof(float), stream);
    hipMemsetAsync(ws.tmsum, 0, CIN * sizeof(float), stream);
    hipMemsetAsync(statsOut, 0, 2 * COUT * sizeof(float), stream);
    segsum_kernel<<<cdiv((long long)nnz * CIN, 256), 256, 0, stream>>>(
        hin, row, col, ws.rsum, ws.csum, nnz, LC);
    seg_finalize_kernel<<<cdiv((long long)n * CIN, 256), 256, 0, stream>>>(
        ws.rsum, ws.csum, ws.rcnt, ws.ccnt, ws.tmsum, n, LC);
    prep_kernel<<<1, 256, 0, stream>>>(W, b, statsPrev, ws.tmsum, ws.cbias,
                                       ws.wEdge, ws.wNode, CIN, COUT, inv_nnz);
    eq_node_kernel<CIN, COUT><<<cdiv(n >> 4, 8), 256, 0, stream>>>(
        ws.rsum, ws.csum, ws.wNode, ws.nodeR, ws.nodeC, n);
    eq_edge_kernel<CIN, COUT><<<cdiv(nnz >> 4, 32), 256, 0, stream>>>(
        hin, row, col, perm, ws.wEdge, ws.nodeR, ws.nodeC, ws.cbias, hout, statsOut, nnz);
}
}  // namespace

extern "C" void kernel_launch(void* const* d_in, const int* in_sizes, int n_in,
                              void* d_out, int out_size, void* d_ws, size_t ws_size,
                              hipStream_t stream) {
    const float* values = (const float*)d_in[0];
    const int*   row    = (const int*)d_in[1];
    const int*   col    = (const int*)d_in[2];
    const int*   perm   = (const int*)d_in[3];
    const float* W1 = (const float*)d_in[4];  const float* b1 = (const float*)d_in[5];
    const float* W2 = (const float*)d_in[6];  const float* b2 = (const float*)d_in[7];
    const float* W3 = (const float*)d_in[8];  const float* b3 = (const float*)d_in[9];
    const float* Wp = (const float*)d_in[10]; const float* bp = (const float*)d_in[11];
    const float* Wl = (const float*)d_in[12]; const float* bl = (const float*)d_in[13];

    const int nnz = in_sizes[1];
    const int n   = out_size / 16;
    const float inv_nnz = 1.f / (float)nnz;

    // carve workspace (256B aligned slices)
    char* cur = (char*)d_ws;
    auto alloc = [&](size_t bytes) -> void* {
        void* r = (void*)cur;
        cur += (bytes + 255) & ~(size_t)255;
        return r;
    };
    Ws ws;
    ws.hA    = (float*)alloc((size_t)nnz * 32 * 4);   // h1, then h3
    ws.hB    = (float*)alloc((size_t)nnz * 64 * 4);   // h2
    ws.rsum  = (float*)alloc((size_t)n * 64 * 4);
    ws.csum  = (float*)alloc((size_t)n * 64 * 4);
    ws.nodeR = (float*)alloc((size_t)n * 64 * 4);
    ws.nodeC = (float*)alloc((size_t)n * 64 * 4);
    ws.rcnt  = (float*)alloc((size_t)n * 4);
    ws.ccnt  = (float*)alloc((size_t)n * 4);
    ws.tmsum = (float*)alloc(64 * 4);
    ws.stats1 = (float*)alloc(2 * 64 * 4);
    ws.stats2 = (float*)alloc(2 * 64 * 4);
    ws.stats3 = (float*)alloc(2 * 64 * 4);
    ws.cbias  = (float*)alloc(64 * 4);
    ws.Wf     = (float*)alloc(512 * 4);
    ws.bfv    = (float*)alloc(16 * 4);
    ws.wEdge  = (__bf16*)alloc(2 * 2 * 4 * 512 * 2);
    ws.wNode  = (__bf16*)alloc(4 * 2 * 4 * 512 * 2);
    if ((size_t)(cur - (char*)d_ws) > ws_size) {
        hipMemsetAsync(d_out, 0, (size_t)out_size * 4, stream);
        return;
    }

    // segment counts (shared by all layers)
    hipMemsetAsync(ws.rcnt, 0, (size_t)n * 4, stream);
    hipMemsetAsync(ws.ccnt, 0, (size_t)n * 4, stream);
    counts_kernel<<<cdiv(nnz, 256), 256, 0, stream>>>(row, col, ws.rcnt, ws.ccnt, nnz);

    // layers (norm folded forward into next layer's weights/bias)
    run_layer<16, 32>(values, W1, b1, nullptr,    ws.stats1, ws.hA, ws, row, col, perm, n, nnz, inv_nnz, stream);
    run_layer<32, 64>(ws.hA,  W2, b2, ws.stats1,  ws.stats2, ws.hB, ws, row, col, perm, n, nnz, inv_nnz, stream);
    run_layer<64, 32>(ws.hB,  W3, b3, ws.stats2,  ws.stats3, ws.hA, ws, row, col, perm, n, nnz, inv_nnz, stream);

    // pooling + heads, with final norm folded into Wp@Wl
    hipMemsetAsync(ws.rsum, 0, (size_t)n * 32 * 4, stream);
    segsum_kernel<<<cdiv((long long)nnz * 32, 256), 256, 0, stream>>>(
        ws.hA, row, col, ws.rsum, nullptr, nnz, 5);
    seg_finalize_kernel<<<cdiv((long long)n * 32, 256), 256, 0, stream>>>(
        ws.rsum, nullptr, ws.rcnt, nullptr, nullptr, n, 5);
    final_prep_kernel<<<1, 256, 0, stream>>>(ws.stats3, Wp, bp, Wl, bl, ws.Wf, ws.bfv, inv_nnz);
    final_out_kernel<<<cdiv(n, 256), 256, 0, stream>>>(ws.rsum, ws.Wf, ws.bfv, (float*)d_out, n);
}
